// Graph_Representation_63110249447920
// MI455X (gfx1250) — compile-verified, matrix operand order checked by reading, not run on hardware
//
#include <hip/hip_runtime.h>
#include <hip/hip_bf16.h>

// ---------------------------------------------------------------------------
// MI455X (gfx1250) implementation.
// conv2 (95% of FLOPs) -> implicit-im2col GEMM on v_wmma_f32_16x16x32_bf16.
// Weights pre-swizzled into per-lane WMMA-B fragment layout (L2-resident).
// A-fragments gathered branch-free as aligned b32 pairs via a per-pair table
// and a one-element-shifted y1 shadow copy (fixes odd-lane alignment).
// Node tiles staged with global_load_async_to_lds_b32 (ASYNCcnt path).
// ---------------------------------------------------------------------------

typedef __attribute__((ext_vector_type(16))) __bf16       v16bf;
typedef __attribute__((ext_vector_type(8)))  float        v8f;
typedef __attribute__((ext_vector_type(4)))  unsigned int v4u;
typedef __attribute__((ext_vector_type(2)))  unsigned int v2u;

#define GDIM 50
// conv2 GEMM dims: M=192 spatial, N=64 (50 padded), K=1824 (100 segs * 18)
#define KSEG 18           // kw padded 17 -> 18 (even; bf16 pairs never straddle)
#define KTOT 1800         // 100 * 18
#define KT   57           // K-tiles of 32
#define NT   4            // 64/16 N-tiles
#define MT   12           // 192/16 M-tiles
#define FRAG_SHORTS 512   // 32 lanes * 16 bf16 per fragment
#define BLOB_SHORTS (KT * NT * FRAG_SHORTS)   // 116736
#define KPAIRS 912        // 1824 / 2 pair-offset table entries

// y1 layout: [10 ch][21 rows][stride 32] + zeroed tail (pad reads, B weight=0)
#define Y1_DATA 6720      // 10*21*32
#define Y1_SIZE 7104      // covers off(max)+base+1 = 7071, rounded

struct SP {
  const float *w1, *b1, *b2, *w3, *b3, *wl, *bl;
  const float *wnd, *bnd;          // null for reduce stack
  const unsigned short *w2b;       // pre-swizzled bf16 B fragments
  int cin;                         // 2 for MPNN, 1 for reduce
};

union FragA { v16bf v; unsigned int q32[8]; };
union FragB { v16bf v; v4u q[2]; };

__device__ __forceinline__ unsigned short f2bf(float f) {
  unsigned int u = __float_as_uint(f);
  u += 0x7FFFu + ((u >> 16) & 1u);          // round-to-nearest-even
  return (unsigned short)(u >> 16);
}

// --- CDNA5 async global->LDS copy (ASYNCcnt-tracked), per bridge doc ---
__device__ __forceinline__ void async_g2l_b32(void* lds, const void* g) {
  unsigned int ldsoff = (unsigned int)(unsigned long long)lds;  // addr[31:0]
  asm volatile("global_load_async_to_lds_b32 %0, %1, off"
               :: "v"(ldsoff), "v"((unsigned long long)g)
               : "memory");
}
__device__ __forceinline__ void wait_async0() {
  asm volatile("s_wait_asynccnt 0" ::: "memory");
}

// ---------------------------------------------------------------------------
// Pack w2 [50,10,10,17] fp32 -> bf16 fragments in exact per-lane WMMA-B layout.
// Slot e=2v+hf maps to K = kt*32 + (v>=4?16:0) + (l>=16?8:0) + 2*(v&3) + hf,
// column n = nt*16 + (l&15).  Pads (kw==17, k>=1800, n>=50) are zero.
// ---------------------------------------------------------------------------
__global__ __launch_bounds__(256) void pack_w2_kernel(const float* __restrict__ w2,
                                                      unsigned short* __restrict__ blob) {
  int idx = blockIdx.x * 256 + threadIdx.x;
  if (idx >= BLOB_SHORTS) return;
  int e    = idx & 15;
  int lane = (idx >> 4) & 31;
  int nt   = (idx >> 9) & 3;
  int kt   = idx >> 11;
  int v = e >> 1, hf = e & 1;
  int koff = ((v >= 4) ? 16 : 0) + ((lane >= 16) ? 8 : 0) + ((v & 3) << 1) + hf;
  int k = kt * 32 + koff;
  int n = nt * 16 + (lane & 15);
  float val = 0.f;
  if (k < KTOT && n < 50) {
    int s = k / KSEG, kw = k - s * KSEG;
    if (kw < 17) {
      int c = s / 10, kh = s - c * 10;
      val = w2[((n * 10 + c) * 10 + kh) * 17 + kw];
    }
  }
  blob[idx] = f2bf(val);
}

__global__ __launch_bounds__(256) void zero_kernel(float* __restrict__ p, int n) {
  int i = blockIdx.x * 256 + threadIdx.x;
  if (i < n) p[i] = 0.f;
}

// ---------------------------------------------------------------------------
// Shared conv stack: input tile -> h[64] in LDS.  256 threads / 8 waves.
// ---------------------------------------------------------------------------
__device__ __forceinline__ void run_stack(const SP& sp,
    const float* __restrict__ nA, const float* __restrict__ nB, float ea,
    float* s_t, unsigned short* s_y1a, unsigned short* s_y1b,
    unsigned short* s_ktab, float* s_y2, float* s_y3, float* s_h) {
  const int tid  = threadIdx.x;
  const int lane = tid & 31;
  const int wave = tid >> 5;
  const int cin  = sp.cin;

  // ---- stage input tile [cin,10,32] via async global->LDS ----
  for (int i = tid; i < 320; i += 256) async_g2l_b32(&s_t[i], &nA[i]);
  if (cin == 2)
    for (int i = tid; i < 320; i += 256) async_g2l_b32(&s_t[320 + i], &nB[i]);

  // ---- per-pair gather table: s_ktab[j] = y1 offset of K-pair k=2j ----
  for (int j = tid; j < KPAIRS; j += 256) {
    int k = 2 * j;
    unsigned short off = 0;                 // k>=1800: any finite data (B=0)
    if (k < KTOT) {
      int s = k / KSEG, kw = k - s * KSEG;  // kw even, <= 16
      int c = s / 10, kh = s - c * 10;
      off = (unsigned short)((c * 21 + kh) * 32 + kw);
    }
    s_ktab[j] = off;
  }
  // ---- zero y1 tails (pad reads must be finite: NaN*0 = NaN in WMMA) ----
  for (int i = tid + Y1_DATA; i < Y1_SIZE; i += 256) s_y1a[i] = 0;
  for (int i = tid + Y1_DATA - 1; i < Y1_SIZE; i += 256) s_y1b[i] = 0;

  wait_async0();
  __syncthreads();

  // ---- conv1: kernel (10,1), pad 10 on H -> y1 [10,21,32] bf16 in LDS ----
  // dual store: s_y1b[j] == s_y1a[j+1] (alignment shadow for odd lanes)
  for (int i = tid; i < 6720; i += 256) {
    int o = i / 672; int r = i % 672; int h1 = r >> 5; int w = r & 31;
    float acc = sp.b1[o];
    for (int c = 0; c < cin; ++c) {
      const float* wrow = sp.w1 + (o * cin + c) * 10;
      const float* trow = s_t + c * 320;
      #pragma unroll
      for (int kh = 0; kh < 10; ++kh) {
        int ri = h1 + kh - 10;
        if (ri >= 0 && ri < 10) acc += wrow[kh] * trow[ri * 32 + w];
      }
    }
    unsigned short bits = f2bf(acc);
    int idx = (o * 21 + h1) * 32 + w;
    s_y1a[idx] = bits;
    if (idx > 0) s_y1b[idx - 1] = bits;
  }
  __syncthreads();

  // ---- conv2 as WMMA GEMM: D[192 x 64] = A(im2col y1)[192 x 1824] * B(w2) ----
  __builtin_prefetch((const void*)sp.w2b, 0, 1);   // weights -> L2/L0
  {
    const int mlane  = lane & 15;     // ow (and D column index)
    const int laneHi = lane >> 4;
    // parity-corrected base: odd lanes read the shifted shadow at base-1
    const unsigned int* yw =
        (const unsigned int*)((mlane & 1) ? s_y1b : s_y1a);
    for (int mt = wave; mt < MT; mt += 8) {        // oh == mt
      v8f acc[NT] = {};
      const unsigned int base2e = (unsigned int)(mt * 32 + (mlane & ~1));
      for (int kt = 0; kt < KT; ++kt) {
        // 8 pair-offsets for this lane: two aligned 8B runs of the table
        const v2u p0 = *(const v2u*)(s_ktab + kt * 16 + laneHi * 4);
        const v2u p1 = *(const v2u*)(s_ktab + kt * 16 + 8 + laneHi * 4);
        FragA a;
        {
          unsigned int w0 = p0[0], w1 = p0[1], w2 = p1[0], w3 = p1[1];
          a.q32[0] = yw[((w0 & 0xFFFFu) + base2e) >> 1];
          a.q32[1] = yw[((w0 >> 16)     + base2e) >> 1];
          a.q32[2] = yw[((w1 & 0xFFFFu) + base2e) >> 1];
          a.q32[3] = yw[((w1 >> 16)     + base2e) >> 1];
          a.q32[4] = yw[((w2 & 0xFFFFu) + base2e) >> 1];
          a.q32[5] = yw[((w2 >> 16)     + base2e) >> 1];
          a.q32[6] = yw[((w3 & 0xFFFFu) + base2e) >> 1];
          a.q32[7] = yw[((w3 >> 16)     + base2e) >> 1];
        }
        const v4u* wb = (const v4u*)sp.w2b + (size_t)(kt * (NT * 32) + lane) * 2;
        #pragma unroll
        for (int nt = 0; nt < NT; ++nt) {           // A fragment reused x4
          FragB b;
          b.q[0] = wb[nt * 64 + 0];
          b.q[1] = wb[nt * 64 + 1];
          acc[nt] = __builtin_amdgcn_wmma_f32_16x16x32_bf16(
              false, a.v, false, b.v, (short)0, acc[nt], false, false);
        }
      }
      // D layout: vgpr r, lanes<16 -> M=r, N=lane; lanes>=16 -> M=8+r, N=lane-16
      const int mbase = mt * 16 + laneHi * 8;
      #pragma unroll
      for (int nt = 0; nt < NT; ++nt) {
        int n = nt * 16 + mlane;
        if (n < 50) {
          float bias = sp.b2[n];
          #pragma unroll
          for (int r = 0; r < 8; ++r)
            s_y2[n * 192 + mbase + r] = acc[nt][r] + bias;   // y2[c][h*16+w]
        }
      }
    }
  }
  __syncthreads();

  // ---- conv3: [5,50,10,8], W-stride 2 -> y3[5,3,5]; wave-parallel dot ----
  {
    int lo = wave * 10;
    int hi = lo + 10; if (hi > 75) hi = 75;
    for (int oi = lo; oi < hi; ++oi) {
      int n3 = oi / 15; int r = oi % 15; int oh3 = r / 5; int ow3 = r % 5;
      float a = 0.f;
      for (int c = lane; c < 50; c += 32) {
        const float* w3p = sp.w3 + (n3 * 50 + c) * 80;
        const float* y2p = s_y2 + c * 192 + oh3 * 16 + ow3 * 2;
        #pragma unroll
        for (int kh = 0; kh < 10; ++kh)
          #pragma unroll
          for (int kw = 0; kw < 8; ++kw)
            a += w3p[kh * 8 + kw] * y2p[kh * 16 + kw];
      }
      #pragma unroll
      for (int off = 16; off > 0; off >>= 1) a += __shfl_down(a, off, 32);
      if (lane == 0) s_y3[oi] = a + sp.b3[n3];
    }
  }
  __syncthreads();

  // ---- linear 75 -> 64 (+ edge attr in slot 64) ----
  for (int j = tid; j < 64; j += 256) {
    float acc = sp.bl[j];
    for (int i = 0; i < 75; ++i) acc += s_y3[i] * sp.wl[i * 64 + j];
    s_h[j] = acc;
  }
  if (tid == 64) s_h[64] = ea;
  __syncthreads();
}

// ---------------------------------------------------------------------------
// One workgroup per (edge, direction). grid = 2E. fwd: idx<E.
// ---------------------------------------------------------------------------
__global__ __launch_bounds__(256) void edge_msg_kernel(
    const float* __restrict__ xcur, const int* __restrict__ ei,
    const float* __restrict__ eattr, SP fp, SP bp,
    float* __restrict__ xnext, int E) {
  __shared__ float          s_t[640];
  __shared__ __align__(16) unsigned short s_y1a[Y1_SIZE];
  __shared__ __align__(16) unsigned short s_y1b[Y1_SIZE];
  __shared__ __align__(16) unsigned short s_ktab[KPAIRS];
  __shared__ float          s_y2[50 * 192];
  __shared__ float          s_y3[80];
  __shared__ float          s_h[66];

  int idx = blockIdx.x;
  bool isFwd = idx < E;
  int e = isFwd ? idx : idx - E;
  const SP sp = isFwd ? fp : bp;
  int src = ei[e], dst = ei[E + e];
  int a = isFwd ? dst : src;     // x_i channel and aggregation node
  int b = isFwd ? src : dst;     // x_j channel

  run_stack(sp, xcur + (size_t)a * 320, xcur + (size_t)b * 320, eattr[e],
            s_t, s_y1a, s_y1b, s_ktab, s_y2, s_y3, s_h);

  // einsum('h,phb->pb') + bnd, scatter-add into aggregation node
  for (int oi = threadIdx.x; oi < 320; oi += 256) {
    int p = oi >> 5;
    float acc = sp.bnd[oi];
    const float* wp = sp.wnd + (size_t)(p * 65) * 32 + (oi & 31);
    for (int i = 0; i < 65; ++i) acc += s_h[i] * wp[i * 32];
    atomicAdd(xnext + (size_t)a * 320 + oi, acc);
  }
}

// ---------------------------------------------------------------------------
// Reduce stack: one workgroup per node, cin=1, writes h[64] to hbuf.
// ---------------------------------------------------------------------------
__global__ __launch_bounds__(256) void node_red_kernel(
    const float* __restrict__ xcur, SP rp, float* __restrict__ hbuf) {
  __shared__ float          s_t[640];
  __shared__ __align__(16) unsigned short s_y1a[Y1_SIZE];
  __shared__ __align__(16) unsigned short s_y1b[Y1_SIZE];
  __shared__ __align__(16) unsigned short s_ktab[KPAIRS];
  __shared__ float          s_y2[50 * 192];
  __shared__ float          s_y3[80];
  __shared__ float          s_h[66];

  int n = blockIdx.x;
  run_stack(rp, xcur + (size_t)n * 320, nullptr, 0.f,
            s_t, s_y1a, s_y1b, s_ktab, s_y2, s_y3, s_h);
  for (int j = threadIdx.x; j < 64; j += 256) hbuf[(size_t)n * 64 + j] = s_h[j];
}

// ---------------------------------------------------------------------------
// out[j] = sum_n sigmoid(h_n . gw[:,j] + gb[j]) * (h_n . fw[:,j] + fb[j])
// ---------------------------------------------------------------------------
__global__ __launch_bounds__(256) void out_kernel(
    const float* __restrict__ hbuf,
    const float* __restrict__ gw, const float* __restrict__ gb,
    const float* __restrict__ fw, const float* __restrict__ fb,
    float* __restrict__ out, int N) {
  __shared__ float red[256];
  int j = blockIdx.x;
  float acc = 0.f;
  for (int n = threadIdx.x; n < N; n += 256) {
    const float* hn = hbuf + (size_t)n * 64;
    float zg = gb[j], zf = fb[j];
    #pragma unroll 8
    for (int i = 0; i < 64; ++i) {
      float h = hn[i];
      zg += h * gw[i * GDIM + j];
      zf += h * fw[i * GDIM + j];
    }
    acc += zf / (1.f + __expf(-zg));
  }
  red[threadIdx.x] = acc;
  __syncthreads();
  for (int s = 128; s > 0; s >>= 1) {
    if (threadIdx.x < s) red[threadIdx.x] += red[threadIdx.x + s];
    __syncthreads();
  }
  if (threadIdx.x == 0) out[j] = red[0];
}

// ---------------------------------------------------------------------------
// d_in order (setup_inputs dict, flattened):
// 0:x 1:edge_attr 2:edge_index
// 3..12:  fwd {w1,b1,w2,b2,w3,b3,wl,bl,wnd,bnd}
// 13..22: bwd {w1,b1,w2,b2,w3,b3,wl,bl,wnd,bnd}
// 23..30: red {w1,b1,w2,b2,w3,b3,wl,bl}
// 31:gm_w 32:gm_b 33:fm_w 34:fm_b
// ---------------------------------------------------------------------------
extern "C" void kernel_launch(void* const* d_in, const int* in_sizes, int n_in,
                              void* d_out, int out_size, void* d_ws, size_t ws_size,
                              hipStream_t stream) {
  (void)n_in; (void)out_size; (void)ws_size;
  const float* x0    = (const float*)d_in[0];
  const float* eattr = (const float*)d_in[1];
  const int*   ei    = (const int*)d_in[2];
  const int E = in_sizes[1];
  const int N = in_sizes[0] / 320;

  auto F = [&](int i) { return (const float*)d_in[i]; };

  // workspace carve-up (16B-aligned blob offsets)
  float* x1   = (float*)d_ws;
  float* x2   = x1 + (size_t)N * 320;
  float* hbuf = x2 + (size_t)N * 320;
  unsigned short* blobF = (unsigned short*)(hbuf + (size_t)N * 64);
  unsigned short* blobB = blobF + BLOB_SHORTS;
  unsigned short* blobR = blobB + BLOB_SHORTS;

  SP fp { F(3),  F(4),  F(6),  F(7),  F(8),  F(9),  F(10), F(11), F(12), blobF, 2 };
  SP bp { F(13), F(14), F(16), F(17), F(18), F(19), F(20), F(21), F(22), blobB, 2 };
  SP rp { F(23), F(24), F(26), F(27), F(28), F(29), F(30), nullptr, nullptr, blobR, 1 };

  const int pgrid = (BLOB_SHORTS + 255) / 256;
  pack_w2_kernel<<<pgrid, 256, 0, stream>>>(F(5),  blobF);
  pack_w2_kernel<<<pgrid, 256, 0, stream>>>(F(15), blobB);
  pack_w2_kernel<<<pgrid, 256, 0, stream>>>(F(25), blobR);

  const int nzb = (N * 320 + 255) / 256;
  // step 0
  zero_kernel<<<nzb, 256, 0, stream>>>(x1, N * 320);
  edge_msg_kernel<<<2 * E, 256, 0, stream>>>(x0, ei, eattr, fp, bp, x1, E);
  // step 1
  zero_kernel<<<nzb, 256, 0, stream>>>(x2, N * 320);
  edge_msg_kernel<<<2 * E, 256, 0, stream>>>(x1, ei, eattr, fp, bp, x2, E);

  node_red_kernel<<<N, 256, 0, stream>>>(x2, rp, hbuf);
  out_kernel<<<GDIM, 256, 0, stream>>>(hbuf, F(31), F(32), F(33), F(34),
                                       (float*)d_out, N);
}
